// MultiHeadAttention_49263274885138
// MI455X (gfx1250) — compile-verified, matrix operand order checked by reading, not run on hardware
//
#include <hip/hip_runtime.h>

// ---------------------------------------------------------------------------
// Types for CDNA5 WMMA (gfx1250, wave32)
// ---------------------------------------------------------------------------
typedef __attribute__((ext_vector_type(16))) __bf16 v16bf;
typedef __attribute__((ext_vector_type(8)))  __bf16 v8bf;
typedef __attribute__((ext_vector_type(8)))  float  v8f;

union V16 { v16bf v; v8bf h[2]; };
union BFU { unsigned short u; __bf16 b; };
union U4  { uint4 q; unsigned short s[8]; };

__device__ __forceinline__ unsigned short bfbits(float f) {
    unsigned u = __float_as_uint(f);
    return (unsigned short)((u + 0x7fffu + ((u >> 16) & 1u)) >> 16);
}
__device__ __forceinline__ __bf16 f2bf(float f) {
    BFU x; x.u = bfbits(f);
    return x.b;
}
__device__ __forceinline__ float bf2f(__bf16 b) {
    BFU x; x.b = b;
    return __uint_as_float(((unsigned)x.u) << 16);
}
__device__ __forceinline__ unsigned packbf2(float lo, float hi) {
    return (unsigned)bfbits(lo) | ((unsigned)bfbits(hi) << 16);
}

__device__ __forceinline__ v8f wmma_bf16(v16bf a, v16bf b, v8f c) {
    // (neg_a, A, neg_b, B, c_mod, C, reuse_a, reuse_b)
    return __builtin_amdgcn_wmma_f32_16x16x32_bf16(false, a, false, b, (short)0, c,
                                                   false, false);
}

#define BD   128
#define ND   256
#define DD   768
#define HD   12
#define RDD  64
#define TE   32
#define TP   40

// ---------------------------------------------------------------------------
// Kernel 1: QKV projection.  X(32768x768) @ W(768x768) + b, output bf16 in
// (B,H,N,RD) layout.  64x64 tiles, 4 waves, WMMA bf16 K=32.
// ---------------------------------------------------------------------------
__global__ __launch_bounds__(128) void qkv_gemm(
    const float* __restrict__ X,
    const float* __restrict__ Wq, const float* __restrict__ bq,
    const float* __restrict__ Wk, const float* __restrict__ bk,
    const float* __restrict__ Wv, const float* __restrict__ bv,
    __bf16* __restrict__ Q, __bf16* __restrict__ K, __bf16* __restrict__ V)
{
    __shared__ __align__(16) __bf16 sA[64 * 32];
    __shared__ __align__(16) __bf16 sW[64 * 32];

    const float* W; const float* bias; __bf16* Outp;
    if (blockIdx.z == 0)      { W = Wq; bias = bq; Outp = Q; }
    else if (blockIdx.z == 1) { W = Wk; bias = bk; Outp = K; }
    else                      { W = Wv; bias = bv; Outp = V; }

    const int tid  = threadIdx.x;
    const int wave = tid >> 5, lane = tid & 31;
    const int r    = lane & 15, half = lane >> 4;
    const int row0 = blockIdx.x * 64;
    const int col0 = blockIdx.y * 64;
    const int m0   = wave * 16;

    v8f acc[4];
    for (int n = 0; n < 4; n++)
        for (int g = 0; g < 8; g++) acc[n][g] = 0.f;

    for (int kk = 0; kk < DD; kk += 32) {
        // stage A tile (64x32 f32 -> bf16): float4 loads, packed uint2 stores
#pragma unroll
        for (int l4 = tid; l4 < 512; l4 += 128) {
            int rr = l4 >> 3, cc = (l4 & 7) * 4;
            float4 x = *(const float4*)&X[(size_t)(row0 + rr) * DD + kk + cc];
            uint2 p; p.x = packbf2(x.x, x.y); p.y = packbf2(x.z, x.w);
            *(uint2*)&sA[rr * 32 + cc] = p;
        }
        // stage W^T tile: coalesced float4 loads, strided b16 stores
#pragma unroll
        for (int l4 = tid; l4 < 512; l4 += 128) {
            int k = l4 >> 4, n4 = (l4 & 15) * 4;
            float4 wv = *(const float4*)&W[(size_t)(kk + k) * DD + col0 + n4];
            unsigned short* sw = (unsigned short*)sW;
            sw[(n4 + 0) * 32 + k] = bfbits(wv.x);
            sw[(n4 + 1) * 32 + k] = bfbits(wv.y);
            sw[(n4 + 2) * 32 + k] = bfbits(wv.z);
            sw[(n4 + 3) * 32 + k] = bfbits(wv.w);
        }
        // prefetch next k-tile while WMMAs run
        if (kk + 32 < DD) {
            __builtin_prefetch(&X[(size_t)(row0 + (tid >> 1)) * DD + kk + 32 + (tid & 1) * 16], 0, 0);
            __builtin_prefetch(&W[(size_t)(kk + 32 + (tid >> 2)) * DD + col0 + (tid & 3) * 16], 0, 0);
        }
        __syncthreads();

        V16 a;
        const __bf16* ap = &sA[(m0 + r) * 32 + half * 8];
        a.h[0] = *(const v8bf*)ap;
        a.h[1] = *(const v8bf*)(ap + 16);
#pragma unroll
        for (int nt = 0; nt < 4; nt++) {
            V16 bb;
            const __bf16* bp = &sW[(nt * 16 + r) * 32 + half * 8];
            bb.h[0] = *(const v8bf*)bp;
            bb.h[1] = *(const v8bf*)(bp + 16);
            acc[nt] = wmma_bf16(a.v, bb.v, acc[nt]);
        }
        __syncthreads();
    }

    for (int nt = 0; nt < 4; nt++) {
#pragma unroll
        for (int g = 0; g < 8; g++) {
            int ml = m0 + g + half * 8;
            int cl = nt * 16 + r;
            int gr = row0 + ml, col = col0 + cl;
            float val = acc[nt][g] + bias[col];
            int bb = gr >> 8, nn = gr & 255;
            int hh = col >> 6, rd = col & 63;
            Outp[(((size_t)bb * HD + hh) * ND + nn) * RDD + rd] = f2bf(val);
        }
    }
}

// ---------------------------------------------------------------------------
// Kernel 2: per-(b,h,row) type-score tables:
//   SQe[i,t]=q_i.Eeq[t], SQp=q.Epq, SKe[j,t]=k_j.Eek[t], SKp=k.Epk
// ---------------------------------------------------------------------------
__global__ __launch_bounds__(160) void typescore_kernel(
    const __bf16* __restrict__ Qbf, const __bf16* __restrict__ Kbf,
    const float* __restrict__ Eeq, const float* __restrict__ Eek,
    const float* __restrict__ Epq, const float* __restrict__ Epk,
    float* __restrict__ SQe, float* __restrict__ SQp,
    float* __restrict__ SKe, float* __restrict__ SKp)
{
    __shared__ float qrow[RDD];
    __shared__ float krow[RDD];

    const int idx = blockIdx.x;
    const int i = idx & 255;
    const int h = (idx >> 8) % HD;
    const int b = idx / (ND * HD);
    const size_t rbase = (((size_t)b * HD + h) * ND + i) * RDD;
    const size_t obase = ((size_t)b * HD + h) * ND + i;
    const int tid = threadIdx.x;

    if (tid < 64)       qrow[tid]      = bf2f(Qbf[rbase + tid]);
    else if (tid < 128) krow[tid - 64] = bf2f(Kbf[rbase + (tid - 64)]);
    __syncthreads();

    if (tid < 32) {
        int t = tid; float s = 0.f;
        const float* e = Eeq + ((size_t)t * HD + h) * RDD;
        for (int d = 0; d < RDD; d++) s += qrow[d] * e[d];
        SQe[obase * TE + t] = s;
    } else if (tid < 72) {
        int t = tid - 32; float s = 0.f;
        const float* e = Epq + ((size_t)t * HD + h) * RDD;
        for (int d = 0; d < RDD; d++) s += qrow[d] * e[d];
        SQp[obase * TP + t] = s;
    } else if (tid < 104) {
        int t = tid - 72; float s = 0.f;
        const float* e = Eek + ((size_t)t * HD + h) * RDD;
        for (int d = 0; d < RDD; d++) s += krow[d] * e[d];
        SKe[obase * TE + t] = s;
    } else if (tid < 144) {
        int t = tid - 104; float s = 0.f;
        const float* e = Epk + ((size_t)t * HD + h) * RDD;
        for (int d = 0; d < RDD; d++) s += krow[d] * e[d];
        SKp[obase * TP + t] = s;
    }
}

// ---------------------------------------------------------------------------
// Kernel 3: fused attention.  One block = (b, h, 16-row i-tile), 128 threads
// (4 waves).  Scores via WMMA, bias gather, softmax, LDS-atomic type pooling,
// P@V via WMMA, type-embedding epilogue.  P never leaves LDS.
// ---------------------------------------------------------------------------
__global__ __launch_bounds__(128) void attn_kernel(
    const __bf16* __restrict__ Qbf, const __bf16* __restrict__ Kbf,
    const __bf16* __restrict__ Vbf,
    const float* __restrict__ SQe, const float* __restrict__ SQp,
    const float* __restrict__ SKe, const float* __restrict__ SKp,
    const int* __restrict__ conn, const int* __restrict__ dist,
    const unsigned char* __restrict__ mask,
    const float* __restrict__ Eev, const float* __restrict__ Epv,
    __bf16* __restrict__ Zbf)
{
    __shared__ __align__(16) __bf16 sQ [16 * 64];   // q tile (i x rd)
    __shared__ __align__(16) __bf16 sKV[64 * 64];   // K strip / V^T strip
    __shared__ __align__(16) float  sS [16 * 256];  // scores -> probs (f32)
    __shared__ __align__(16) __bf16 sP [16 * 256];  // probs bf16 for WMMA
    __shared__ float sSQ[16 * 72];                  // per-row q-side biases
    __shared__ float sAcc[16 * 72];                 // pooled Ae(32)|Ap(40)
    __shared__ float red[16 * 8];

    const int tid  = threadIdx.x;
    const int wave = tid >> 5, lane = tid & 31;
    const int r    = lane & 15, half = lane >> 4;
    const int i0   = blockIdx.x * 16;
    const int h    = blockIdx.y;
    const int b    = blockIdx.z;
    const size_t bh = (size_t)b * HD + h;
    const size_t qkv_base = (bh * ND) * RDD;

    // ---- P0: stage q tile, q-side bias tables, zero pooling accum ----
    {
        const uint4* src = (const uint4*)(Qbf + qkv_base + (size_t)i0 * RDD);
        uint4* dst = (uint4*)sQ;
        for (int l = tid; l < 128; l += 128) dst[l] = src[l];
    }
    for (int l = tid; l < 16 * 72; l += 128) {
        int i = l / 72, c = l % 72;
        float v;
        if (c < TE) v = SQe[(bh * ND + i0 + i) * TE + c];
        else        v = SQp[(bh * ND + i0 + i) * TP + (c - TE)];
        sSQ[l] = v;
    }
    for (int l = tid; l < 16 * 72; l += 128) sAcc[l] = 0.f;
    __syncthreads();

    // ---- P1: raw scores  S = q @ k^T  (16 x 256), strips of 64 j ----
    for (int js = 0; js < ND; js += 64) {
        {
            const uint4* src = (const uint4*)(Kbf + qkv_base + (size_t)js * RDD);
            uint4* dst = (uint4*)sKV;
#pragma unroll
            for (int l = tid; l < 512; l += 128) dst[l] = src[l];
        }
        if (js + 64 < ND)
            __builtin_prefetch(&Kbf[qkv_base + (size_t)(js + 64) * RDD + tid * 32], 0, 0);
        __syncthreads();

        v8f acc;
        for (int g = 0; g < 8; g++) acc[g] = 0.f;
#pragma unroll
        for (int ks = 0; ks < 2; ks++) {
            V16 a, bb;
            const __bf16* ap = &sQ[r * 64 + ks * 32 + half * 8];
            a.h[0] = *(const v8bf*)ap;
            a.h[1] = *(const v8bf*)(ap + 16);
            const __bf16* bp = &sKV[(wave * 16 + r) * 64 + ks * 32 + half * 8];
            bb.h[0] = *(const v8bf*)bp;
            bb.h[1] = *(const v8bf*)(bp + 16);
            acc = wmma_bf16(a.v, bb.v, acc);
        }
#pragma unroll
        for (int g = 0; g < 8; g++) {
            int i = g + half * 8;
            int j = js + wave * 16 + r;
            sS[i * 256 + j] = acc[g];
        }
        __syncthreads();
    }

    // ---- P2: add type biases, scale, mask ----
    const size_t tybase = ((size_t)b * ND + i0) * ND;
    for (int e = tid; e < 4096; e += 128) {
        int i = e >> 8, j = e & 255;
        int te = conn[tybase + (size_t)i * ND + j];
        int tp = dist[tybase + (size_t)i * ND + j];
        float v = sS[e] + sSQ[i * 72 + te] + sSQ[i * 72 + TE + tp]
                + SKe[(bh * ND + j) * TE + te]
                + SKp[(bh * ND + j) * TP + tp];
        v *= 0.125f;                       // RD^-0.5
        if (mask[b * ND + j]) v = -__builtin_inff();
        sS[e] = v;
    }
    __syncthreads();

    // ---- P3: softmax over j (8 threads per row) ----
    {
        const int row = tid >> 3, sub = tid & 7;
        float m = -__builtin_inff();
        for (int j = sub * 32; j < sub * 32 + 32; j++)
            m = fmaxf(m, sS[row * 256 + j]);
        red[row * 8 + sub] = m;
        __syncthreads();
        float rm = -__builtin_inff();
        for (int s = 0; s < 8; s++) rm = fmaxf(rm, red[row * 8 + s]);
        float psum = 0.f;
        for (int j = sub * 32; j < sub * 32 + 32; j++) {
            float p = __expf(sS[row * 256 + j] - rm);
            sS[row * 256 + j] = p;
            psum += p;
        }
        __syncthreads();
        red[row * 8 + sub] = psum;
        __syncthreads();
        float rs = 0.f;
        for (int s = 0; s < 8; s++) rs += red[row * 8 + s];
        float inv = 1.f / rs;
        for (int j = sub * 32; j < sub * 32 + 32; j++) {
            float p = sS[row * 256 + j] * inv;
            sS[row * 256 + j] = p;
            sP[row * 256 + j] = f2bf(p);
        }
    }
    __syncthreads();

    // ---- P4: per-type pooling of probabilities (LDS atomics) ----
    for (int e = tid; e < 4096; e += 128) {
        int i = e >> 8, j = e & 255;
        int te = conn[tybase + (size_t)i * ND + j];
        int tp = dist[tybase + (size_t)i * ND + j];
        float p = sS[e];
        atomicAdd(&sAcc[i * 72 + te], p);
        atomicAdd(&sAcc[i * 72 + TE + tp], p);
    }
    __syncthreads();

    // ---- P5: Z = P @ V  + type-embedding epilogue ----
    v8f zacc;
    for (int g = 0; g < 8; g++) zacc[g] = 0.f;
    for (int js = 0; js < ND; js += 64) {
        // stage V^T strip: uint4 loads (8 bf16 along rd), strided b16 stores
#pragma unroll
        for (int l = tid; l < 512; l += 128) {
            int jl = l >> 3, rc = (l & 7) * 8;
            U4 u;
            u.q = *(const uint4*)&Vbf[qkv_base + (size_t)(js + jl) * RDD + rc];
            unsigned short* skv = (unsigned short*)sKV;
#pragma unroll
            for (int t = 0; t < 8; t++) skv[(rc + t) * 64 + jl] = u.s[t];
        }
        if (js + 64 < ND)
            __builtin_prefetch(&Vbf[qkv_base + (size_t)(js + 64) * RDD + tid * 32], 0, 0);
        __syncthreads();
#pragma unroll
        for (int ks = 0; ks < 2; ks++) {
            V16 a, bb;
            const __bf16* ap = &sP[r * 256 + js + ks * 32 + half * 8];
            a.h[0] = *(const v8bf*)ap;
            a.h[1] = *(const v8bf*)(ap + 16);
            const __bf16* bp = &sKV[(wave * 16 + r) * 64 + ks * 32 + half * 8];
            bb.h[0] = *(const v8bf*)bp;
            bb.h[1] = *(const v8bf*)(bp + 16);
            zacc = wmma_bf16(a.v, bb.v, zacc);
        }
        __syncthreads();
    }

#pragma unroll
    for (int g = 0; g < 8; g++) {
        int i  = g + half * 8;
        int rd = wave * 16 + r;
        float ze = 0.f;
        for (int t = 0; t < TE; t++)
            ze += sAcc[i * 72 + t] * Eev[((size_t)t * HD + h) * RDD + rd];
        for (int t = 0; t < TP; t++)
            ze += sAcc[i * 72 + TE + t] * Epv[((size_t)t * HD + h) * RDD + rd];
        float z = zacc[g] + ze;
        Zbf[((size_t)b * ND + i0 + i) * DD + h * RDD + rd] = f2bf(z);
    }
}

// ---------------------------------------------------------------------------
// Kernel 4: output projection  Z(32768x768,bf16) @ Wo + bo -> f32 out
// ---------------------------------------------------------------------------
__global__ __launch_bounds__(128) void out_gemm(
    const __bf16* __restrict__ Z, const float* __restrict__ W,
    const float* __restrict__ bias, float* __restrict__ Out)
{
    __shared__ __align__(16) __bf16 sA[64 * 32];
    __shared__ __align__(16) __bf16 sW[64 * 32];

    const int tid  = threadIdx.x;
    const int wave = tid >> 5, lane = tid & 31;
    const int r    = lane & 15, half = lane >> 4;
    const int row0 = blockIdx.x * 64;
    const int col0 = blockIdx.y * 64;
    const int m0   = wave * 16;

    v8f acc[4];
    for (int n = 0; n < 4; n++)
        for (int g = 0; g < 8; g++) acc[n][g] = 0.f;

    for (int kk = 0; kk < DD; kk += 32) {
        // stage A (bf16 copy): uint4 bulk copy
#pragma unroll
        for (int l = tid; l < 256; l += 128) {
            int rr = l >> 2, c8 = (l & 3) * 8;
            *(uint4*)&sA[rr * 32 + c8] =
                *(const uint4*)&Z[(size_t)(row0 + rr) * DD + kk + c8];
        }
        // stage W^T tile
#pragma unroll
        for (int l4 = tid; l4 < 512; l4 += 128) {
            int k = l4 >> 4, n4 = (l4 & 15) * 4;
            float4 wv = *(const float4*)&W[(size_t)(kk + k) * DD + col0 + n4];
            unsigned short* sw = (unsigned short*)sW;
            sw[(n4 + 0) * 32 + k] = bfbits(wv.x);
            sw[(n4 + 1) * 32 + k] = bfbits(wv.y);
            sw[(n4 + 2) * 32 + k] = bfbits(wv.z);
            sw[(n4 + 3) * 32 + k] = bfbits(wv.w);
        }
        if (kk + 32 < DD) {
            __builtin_prefetch(&Z[(size_t)(row0 + (tid >> 1)) * DD + kk + 32 + (tid & 1) * 16], 0, 0);
            __builtin_prefetch(&W[(size_t)(kk + 32 + (tid >> 2)) * DD + col0 + (tid & 3) * 16], 0, 0);
        }
        __syncthreads();

        V16 a;
        const __bf16* ap = &sA[(m0 + r) * 32 + half * 8];
        a.h[0] = *(const v8bf*)ap;
        a.h[1] = *(const v8bf*)(ap + 16);
#pragma unroll
        for (int nt = 0; nt < 4; nt++) {
            V16 bb;
            const __bf16* bp = &sW[(nt * 16 + r) * 32 + half * 8];
            bb.h[0] = *(const v8bf*)bp;
            bb.h[1] = *(const v8bf*)(bp + 16);
            acc[nt] = wmma_bf16(a.v, bb.v, acc[nt]);
        }
        __syncthreads();
    }

    for (int nt = 0; nt < 4; nt++) {
#pragma unroll
        for (int g = 0; g < 8; g++) {
            int ml = m0 + g + half * 8;
            int cl = nt * 16 + r;
            int gr = row0 + ml, col = col0 + cl;
            Out[(size_t)gr * DD + col] = acc[nt][g] + bias[col];
        }
    }
}

// ---------------------------------------------------------------------------
// Host launcher
// ---------------------------------------------------------------------------
extern "C" void kernel_launch(void* const* d_in, const int* in_sizes, int n_in,
                              void* d_out, int out_size, void* d_ws, size_t ws_size,
                              hipStream_t stream)
{
    (void)in_sizes; (void)n_in; (void)out_size; (void)ws_size;

    const float* X            = (const float*)d_in[0];
    const int* conn           = (const int*)d_in[1];
    const int* dist           = (const int*)d_in[2];
    const unsigned char* mask = (const unsigned char*)d_in[3];
    const float* Wq = (const float*)d_in[4];  const float* bq = (const float*)d_in[5];
    const float* Wk = (const float*)d_in[6];  const float* bk = (const float*)d_in[7];
    const float* Wv = (const float*)d_in[8];  const float* bv = (const float*)d_in[9];
    const float* Wo = (const float*)d_in[10]; const float* bo = (const float*)d_in[11];
    const float* Eeq = (const float*)d_in[12];
    const float* Eek = (const float*)d_in[13];
    const float* Eev = (const float*)d_in[14];
    const float* Epq = (const float*)d_in[15];
    const float* Epk = (const float*)d_in[16];
    const float* Epv = (const float*)d_in[17];
    float* Out = (float*)d_out;

    const size_t BHN   = (size_t)BD * HD * ND;     // 393216
    const size_t BHNRD = BHN * RDD;                // 25.2M

    char* ws = (char*)d_ws;
    size_t off = 0;
    auto alloc = [&](size_t bytes) -> char* {
        char* p = ws + off;
        off += (bytes + 255) & ~(size_t)255;
        return p;
    };
    __bf16* Qbf = (__bf16*)alloc(BHNRD * 2);
    __bf16* Kbf = (__bf16*)alloc(BHNRD * 2);
    __bf16* Vbf = (__bf16*)alloc(BHNRD * 2);
    float*  SQe = (float*)alloc(BHN * TE * 4);
    float*  SQp = (float*)alloc(BHN * TP * 4);
    float*  SKe = (float*)alloc(BHN * TE * 4);
    float*  SKp = (float*)alloc(BHN * TP * 4);
    __bf16* Zbf = (__bf16*)alloc((size_t)BD * ND * DD * 2);

    qkv_gemm<<<dim3(512, 12, 3), 128, 0, stream>>>(
        X, Wq, bq, Wk, bk, Wv, bv, Qbf, Kbf, Vbf);

    typescore_kernel<<<dim3((unsigned)BHN), 160, 0, stream>>>(
        Qbf, Kbf, Eeq, Eek, Epq, Epk, SQe, SQp, SKe, SKp);

    attn_kernel<<<dim3(16, 12, 128), 128, 0, stream>>>(
        Qbf, Kbf, Vbf, SQe, SQp, SKe, SKp, conn, dist, mask, Eev, Epv, Zbf);

    out_gemm<<<dim3(512, 12, 1), 128, 0, stream>>>(Zbf, Wo, bo, Out);
}
